// MoEFFN_11295763988746
// MI455X (gfx1250) — matrix operand order repaired and checked by reading, not compiled, not run on hardware
//
#include <hip/hip_runtime.h>
#include <hip/hip_bf16.h>
#include <math.h>

// ---------------- problem constants ----------------
#define T_TOK   8192      // B*S tokens
#define DDIM    1024
#define FDIM    4096
#define NEXP    8

// ---------------- tiling ----------------
#define MT      128       // tokens per tile
#define NT      128       // output dims per tile
#define FC      64        // F chunk
#define KC      64        // K chunk over D
#define XS_S    72        // padded LDS strides (bf16 elems, 144B rows -> 16B aligned)
#define W1T_S   72        // W1 staged transposed: [f][k]
#define HS_S    72
#define W2T_S   72        // W2 staged transposed: [d][f]
#define MAXTILE (T_TOK / MT)   // 64: worst case one expert owns every token

typedef __attribute__((ext_vector_type(16))) __bf16 v16bf;
typedef __attribute__((ext_vector_type(8)))  __bf16 v8bf;
typedef __attribute__((ext_vector_type(8)))  float  v8f;

struct V16pair { v8bf lo, hi; };

static __device__ inline __bf16 f2bf(float f) { return (__bf16)f; }

static __device__ inline v8f v8f_zero() {
  v8f z;
#pragma unroll
  for (int i = 0; i < 8; ++i) z[i] = 0.f;
  return z;
}

// A fragment: 16(M) x 32(K) bf16 from [row][k] layout, 16B-aligned rows.
// ISA 7.12.2: lane<16 -> M=lane, elems {K=kb..kb+7, K=16+kb..16+kb+7}, kb=8*(lane>=16).
static __device__ inline v16bf load_a(const __bf16* p, int stride) {
  const int lane = threadIdx.x & 31;
  const int m  = lane & 15;
  const int kb = (lane >> 4) << 3;
  const __bf16* q = p + m * stride + kb;
  const v8bf lo = *(const v8bf*)q;          // K = kb .. kb+7
  const v8bf hi = *(const v8bf*)(q + 16);   // K = 16+kb .. 16+kb+7
  return __builtin_bit_cast(v16bf, V16pair{lo, hi});
}

// B fragment: 32(K) x 16(N) bf16 from TRANSPOSED [n][k] layout.
// lane<16 -> N=lane, K={kb..kb+7, 16+kb..16+kb+7}, kb=8*(lane>=16).
static __device__ inline v16bf load_bt(const __bf16* p, int stride) {
  const int lane = threadIdx.x & 31;
  const int n  = lane & 15;
  const int kb = (lane >> 4) << 3;
  const __bf16* q = p + n * stride + kb;
  const v8bf lo = *(const v8bf*)q;
  const v8bf hi = *(const v8bf*)(q + 16);
  return __builtin_bit_cast(v16bf, V16pair{lo, hi});
}

static __device__ inline void wait_asynccnt0() {
#if __has_builtin(__builtin_amdgcn_s_wait_asynccnt)
  __builtin_amdgcn_s_wait_asynccnt(0);
#else
  asm volatile("s_wait_asynccnt 0" ::: "memory");
#endif
}

// ---------------- kernel 1: fp32 -> bf16 weight conversion ----------------
__global__ __launch_bounds__(256)
void moe_convw_kernel(const float* __restrict__ src, __bf16* __restrict__ dst) {
  const size_t i = ((size_t)blockIdx.x * 256 + threadIdx.x) * 8;
#pragma unroll
  for (int j = 0; j < 8; ++j) dst[i + j] = f2bf(src[i + j]);
}

// ---------------- kernel 2: router (one wave per token) ----------------
__global__ __launch_bounds__(256)
void moe_router_kernel(const float* __restrict__ x,
                       const float* __restrict__ rw,   // [D, E]
                       const float* __restrict__ rb,   // [E]
                       __bf16* __restrict__ xb,        // [T, D] bf16 out
                       float* __restrict__ wts,        // [T, 2] renorm top-2 probs
                       int*   __restrict__ lists,      // [E, T] pair ids (t<<1 | slot)
                       int*   __restrict__ counts)     // [E]
{
  const int lane = threadIdx.x & 31;
  const int wid  = threadIdx.x >> 5;
  const int t = blockIdx.x * 8 + wid;

  float acc[NEXP];
#pragma unroll
  for (int e = 0; e < NEXP; ++e) acc[e] = 0.f;

  for (int i = 0; i < DDIM / 32; ++i) {
    const int d = i * 32 + lane;
    const float xv = x[(size_t)t * DDIM + d];
    xb[(size_t)t * DDIM + d] = f2bf(xv);
#pragma unroll
    for (int e = 0; e < NEXP; ++e) acc[e] += xv * rw[d * NEXP + e];
  }
#pragma unroll
  for (int e = 0; e < NEXP; ++e) {
#pragma unroll
    for (int m = 16; m >= 1; m >>= 1) acc[e] += __shfl_xor(acc[e], m, 32);
  }

  if (lane == 0) {
    float lg[NEXP], p[NEXP];
    float mx = -3.0e38f;
#pragma unroll
    for (int e = 0; e < NEXP; ++e) { lg[e] = acc[e] + rb[e]; mx = fmaxf(mx, lg[e]); }
    float s = 0.f;
#pragma unroll
    for (int e = 0; e < NEXP; ++e) { p[e] = __expf(lg[e] - mx); s += p[e]; }
    const float inv = 1.f / s;
#pragma unroll
    for (int e = 0; e < NEXP; ++e) p[e] *= inv;
    int i0 = 0;
#pragma unroll
    for (int e = 1; e < NEXP; ++e) if (p[e] > p[i0]) i0 = e;
    int i1 = (i0 == 0) ? 1 : 0;
#pragma unroll
    for (int e = 0; e < NEXP; ++e) if (e != i0 && p[e] > p[i1]) i1 = e;
    const float s2 = p[i0] + p[i1];
    wts[t * 2 + 0] = p[i0] / s2;
    wts[t * 2 + 1] = p[i1] / s2;
    const int pos0 = atomicAdd(&counts[i0], 1);
    lists[i0 * T_TOK + pos0] = (t << 1);
    const int pos1 = atomicAdd(&counts[i1], 1);
    lists[i1 * T_TOK + pos1] = (t << 1) | 1;
  }
}

// ---------------- kernel 3: grouped expert FFN (WMMA) ----------------
__global__ __launch_bounds__(256)
void moe_expert_kernel(const __bf16* __restrict__ xb,
                       const __bf16* __restrict__ w1b,  // [E, D, F] bf16
                       const __bf16* __restrict__ w2b,  // [E, F, D] bf16
                       const float*  __restrict__ b1,   // [E, F]
                       const float*  __restrict__ b2,   // [E, D]
                       const float*  __restrict__ wts,
                       const int*    __restrict__ lists,
                       const int*    __restrict__ counts,
                       float* __restrict__ y_ws)        // [2, T, D]
{
  __shared__ __align__(16) __bf16 XS[MT * XS_S];    // [token][k]
  __shared__ __align__(16) __bf16 W1S[FC * W1T_S];  // transposed: [f][k]
  __shared__ __align__(16) __bf16 HS[MT * HS_S];    // [token][f]
  __shared__ __align__(16) __bf16 W2S[NT * W2T_S];  // transposed: [d][f]
  __shared__ int   tokS[MT];
  __shared__ int   slotS[MT];
  __shared__ float wgtS[MT];

  const int e    = blockIdx.x / MAXTILE;
  const int tile = blockIdx.x % MAXTILE;
  const int ntb  = blockIdx.y;            // 128-wide slab of D
  const int cnt  = counts[e];
  const int base = tile * MT;
  if (base >= cnt) return;                // uniform across block

  const int tid  = threadIdx.x;
  const int lane = tid & 31;
  const int wid  = tid >> 5;
  const int wm   = wid >> 1;              // 0..3 (row group of 32)
  const int wn   = wid & 1;               // 0..1 (col group of 64)

  if (tid < MT) {
    const int idx = base + tid;
    int t = 0, s = 0; float w = 0.f;
    if (idx < cnt) {
      const int v = lists[e * T_TOK + idx];
      t = v >> 1; s = v & 1; w = wts[t * 2 + s];
    }
    tokS[tid] = t; slotS[tid] = s; wgtS[tid] = w;
  }
  __syncthreads();

  v8f acc_y[8];
#pragma unroll
  for (int i = 0; i < 8; ++i) acc_y[i] = v8f_zero();

  const __bf16* w1e = w1b + (size_t)e * DDIM * FDIM;
  const __bf16* w2e = w2b + (size_t)e * FDIM * DDIM;

  for (int fc = 0; fc < FDIM / FC; ++fc) {
    // ---- h_chunk = x_tile[128,D] @ w1[:, fc*64 .. +64] ----
    v8f acc_h[4];
#pragma unroll
    for (int i = 0; i < 4; ++i) acc_h[i] = v8f_zero();

    for (int kc = 0; kc < DDIM / KC; ++kc) {
      __syncthreads();   // prior readers of XS/W1S done

      // X tile: async DMA global -> LDS, 16B chunks (row layout preserved).
      for (int c = tid; c < MT * (KC / 8); c += 256) {   // 1024 chunks of 8 bf16
        const int r = c >> 3, q = (c & 7) * 8;
        const unsigned lds = (unsigned)(uintptr_t)&XS[r * XS_S + q];
        const __bf16* g = xb + (size_t)tokS[r] * DDIM + kc * KC + q;
        asm volatile("global_load_async_to_lds_b128 %0, %1, off"
                     :: "v"(lds), "v"(g) : "memory");
      }
      // W1 chunk: coalesced global read, transposed LDS write [f][k].
      for (int i = tid; i < KC * FC; i += 256) {
        const int k = i >> 6, f = i & 63;
        W1S[f * W1T_S + k] = w1e[(size_t)(kc * KC + k) * FDIM + fc * FC + f];
      }
      if (kc + 1 < DDIM / KC)
        __builtin_prefetch(&w1e[(size_t)((kc + 1) * KC) * FDIM + fc * FC + tid], 0, 0);
      wait_asynccnt0();
      __syncthreads();

#pragma unroll
      for (int ks = 0; ks < 2; ++ks) {
        const v16bf a = load_a(&XS[(wid * 16) * XS_S + ks * 32], XS_S);
#pragma unroll
        for (int ft = 0; ft < 4; ++ft) {
          const v16bf b = load_bt(&W1S[(ft * 16) * W1T_S + ks * 32], W1T_S);
          acc_h[ft] = __builtin_amdgcn_wmma_f32_16x16x32_bf16(
              false, a, false, b, (short)0, acc_h[ft], false, false);
        }
      }
    }

    // ---- bias + exact gelu + per-row combine-weight scale -> HS (bf16) ----
#pragma unroll
    for (int ft = 0; ft < 4; ++ft) {
      const int col = ft * 16 + (lane & 15);
      const float b1v = b1[e * FDIM + fc * FC + col];
#pragma unroll
      for (int r = 0; r < 8; ++r) {
        const int row = wid * 16 + r + ((lane >> 4) << 3);
        const float v = acc_h[ft][r] + b1v;
        const float g = 0.5f * v * (1.0f + erff(v * 0.70710678118654752f));
        HS[row * HS_S + col] = f2bf(g * wgtS[row]);
      }
    }
    // ---- stage w2 chunk, transposed [d][f] ----
    for (int i = tid; i < FC * NT; i += 256) {
      const int f = i >> 7, d = i & 127;
      W2S[d * W2T_S + f] = w2e[(size_t)(fc * FC + f) * DDIM + ntb * NT + d];
    }
    __syncthreads();   // HS + W2S visible to all waves

    // ---- acc_y += (cw*gelu(h)) @ w2_chunk ----
#pragma unroll
    for (int ks = 0; ks < 2; ++ks) {
#pragma unroll
      for (int mi = 0; mi < 2; ++mi) {
        const v16bf a = load_a(&HS[(wm * 32 + mi * 16) * HS_S + ks * 32], HS_S);
#pragma unroll
        for (int ni = 0; ni < 4; ++ni) {
          const v16bf b = load_bt(&W2S[(wn * 64 + ni * 16) * W2T_S + ks * 32], W2T_S);
          acc_y[mi * 4 + ni] = __builtin_amdgcn_wmma_f32_16x16x32_bf16(
              false, a, false, b, (short)0, acc_y[mi * 4 + ni], false, false);
        }
      }
    }
  }

  // ---- store per-(token,slot): cw*(h@w2) + cw*b2 ----
  int nvalid = cnt - base; if (nvalid > MT) nvalid = MT;
#pragma unroll
  for (int mi = 0; mi < 2; ++mi) {
#pragma unroll
    for (int ni = 0; ni < 4; ++ni) {
#pragma unroll
      for (int r = 0; r < 8; ++r) {
        const int row = wm * 32 + mi * 16 + r + ((lane >> 4) << 3);
        if (row < nvalid) {
          const int t = tokS[row], s = slotS[row];
          const int d = ntb * NT + wn * 64 + ni * 16 + (lane & 15);
          y_ws[((size_t)s * T_TOK + t) * DDIM + d] =
              acc_y[mi * 4 + ni][r] + wgtS[row] * b2[e * DDIM + d];
        }
      }
    }
  }
}

// ---------------- kernel 4: combine slot0 + slot1 ----------------
__global__ __launch_bounds__(256)
void moe_combine_kernel(const float* __restrict__ y, float* __restrict__ out) {
  const size_t i = ((size_t)blockIdx.x * 256 + threadIdx.x) * 4;
#pragma unroll
  for (int j = 0; j < 4; ++j)
    out[i + j] = y[i + j] + y[(size_t)T_TOK * DDIM + i + j];
}

// ---------------- launcher ----------------
extern "C" void kernel_launch(void* const* d_in, const int* in_sizes, int n_in,
                              void* d_out, int out_size, void* d_ws, size_t ws_size,
                              hipStream_t stream) {
  const float* x  = (const float*)d_in[0];
  const float* w1 = (const float*)d_in[1];
  const float* b1 = (const float*)d_in[2];
  const float* w2 = (const float*)d_in[3];
  const float* b2 = (const float*)d_in[4];
  const float* rw = (const float*)d_in[5];
  const float* rb = (const float*)d_in[6];
  float* out = (float*)d_out;

  char* ws = (char*)d_ws;
  size_t off = 0;
  int*    counts = (int*)(ws + off);    off += 256;
  int*    lists  = (int*)(ws + off);    off += (size_t)NEXP * T_TOK * sizeof(int);
  float*  wts    = (float*)(ws + off);  off += (size_t)T_TOK * 2 * sizeof(float);
  __bf16* xb     = (__bf16*)(ws + off); off += (size_t)T_TOK * DDIM * 2;
  __bf16* w1b    = (__bf16*)(ws + off); off += (size_t)NEXP * DDIM * FDIM * 2;
  __bf16* w2b    = (__bf16*)(ws + off); off += (size_t)NEXP * FDIM * DDIM * 2;
  float*  y_ws   = (float*)(ws + off);  off += (size_t)2 * T_TOK * DDIM * sizeof(float);

  hipMemsetAsync(counts, 0, 256, stream);

  const size_t nw = (size_t)NEXP * DDIM * FDIM;          // 33.5M elems each
  moe_convw_kernel<<<dim3((unsigned)(nw / (256 * 8))), 256, 0, stream>>>(w1, w1b);
  moe_convw_kernel<<<dim3((unsigned)(nw / (256 * 8))), 256, 0, stream>>>(w2, w2b);

  moe_router_kernel<<<dim3(T_TOK / 8), 256, 0, stream>>>(x, rw, rb, xb, wts, lists, counts);

  moe_expert_kernel<<<dim3(NEXP * MAXTILE, DDIM / NT), 256, 0, stream>>>(
      xb, w1b, w2b, b1, b2, wts, lists, counts, y_ws);

  moe_combine_kernel<<<dim3(T_TOK * DDIM / (256 * 4)), 256, 0, stream>>>(y_ws, out);
}
